// SparseInputPerceiver_78460462563524
// MI455X (gfx1250) — matrix operand order repaired and checked by reading, not compiled, not run on hardware
//
#include <hip/hip_runtime.h>
#include <math.h>

// ---------------- types / helpers ----------------
typedef __attribute__((ext_vector_type(16))) __bf16          v16bf;
typedef __attribute__((ext_vector_type(16))) unsigned short  v16u;
typedef __attribute__((ext_vector_type(8)))  float           v8f;
typedef __attribute__((ext_vector_type(4)))  int             v4i;

#define DEVFN __device__ __forceinline__

#if __has_builtin(__builtin_amdgcn_global_load_async_to_lds_b128) && \
    __has_builtin(__builtin_amdgcn_s_wait_asynccnt)
#define HAVE_ASYNC_LDS 1
typedef __attribute__((address_space(1))) v4i* gptr_v4i;
typedef __attribute__((address_space(3))) v4i* lptr_v4i;
#else
#define HAVE_ASYNC_LDS 0
#endif

DEVFN unsigned short f2bf(float f) {
  unsigned u = __builtin_bit_cast(unsigned, f);
  return (unsigned short)((u + 0x7FFFu + ((u >> 16) & 1u)) >> 16);
}
DEVFN float bf2f(unsigned short h) {
  unsigned u = ((unsigned)h) << 16;
  return __builtin_bit_cast(float, u);
}

// model dims
#define NB     4
#define NTB    8
#define NVOX   8192
#define IDW    435
#define IDWP   448    // padded (mult 32)
#define NL     128
#define LD     512
#define INW    600
#define INWP   608    // padded attn-out stride (mult 32)
#define INWN   640    // padded q/o weight N (mult 64)
#define KVW    1200
#define KVWN   1216   // padded kv weight N (mult 64)
#define FFW    2048
#define OCW    64
#define NHEAD  8
#define DH     75
#define DP     80     // Vt rows
#define DPP    96     // padded head dim for Q/K tiles (3 x 32)
#define KT     64     // key tile
#define DEPTH  5

// ============================================================
// build data: patchify + mask + fourier encode + unit-normalize
// one wave per voxel; output stride IDWP (pad pre-zeroed)
// ============================================================
__global__ __launch_bounds__(128)
void k_build_data(const float* __restrict__ input,
                  unsigned short* __restrict__ dataN,
                  float* __restrict__ maskb)
{
  __shared__ float dbuf[4][448];
  int tid = threadIdx.x, lane = tid & 31, wave = tid >> 5;
  int g = blockIdx.x * 4 + wave;
  int b = g >> 13;
  int n = g & (NVOX - 1);
  int y = n >> 8;
  int x = (n >> 3) & 31;
  int t = n & 7;
  for (int q = lane; q < 48; q += 32) {
    int c = q % 3, pp = q / 3;
    int py = pp >> 2, px = pp & 3;
    int Y = y * 4 + py, X = x * 4 + px;
    dbuf[wave][q] = input[((((size_t)t * NB + b) * 3 + c) * 128 + Y) * 128 + X];
  }
  float mx = -3.0e38f;
  for (int q = lane; q < 48; q += 32) mx = fmaxf(mx, dbuf[wave][q]);
  #pragma unroll
  for (int m = 16; m >= 1; m >>= 1) mx = fmaxf(mx, __shfl_xor(mx, m, 32));
  float maskf = (fabsf(mx) > 0.3f) ? 1.f : 0.f;
  float pos0 = (2.f * y / 32.f - 1.f) * maskf;
  float pos1 = (2.f * x / 32.f - 1.f) * maskf;
  float pos2 = (2.f * t / 8.f  - 1.f) * maskf;
  float ss = 0.f;
  for (int e = lane; e < IDW; e += 32) {
    float v;
    if (e < 48) v = dbuf[wave][e] * maskf;
    else {
      int e2 = e - 48;
      int j = e2 / 129, r = e2 % 129;
      float pj = (j == 0) ? pos0 : (j == 1 ? pos1 : pos2);
      if (r < 64)       v = __sinf(ldexpf(pj, -r));
      else if (r < 128) v = __cosf(ldexpf(pj, -(r - 64)));
      else              v = pj;
    }
    dbuf[wave][e] = v;
    ss += v * v;
  }
  #pragma unroll
  for (int m = 16; m >= 1; m >>= 1) ss += __shfl_xor(ss, m, 32);
  float inv = 1.f / fmaxf(sqrtf(ss), 1e-5f);
  for (int e = lane; e < IDW; e += 32)
    dataN[(size_t)g * IDWP + e] = f2bf(dbuf[wave][e] * inv);
  if (lane == 0) maskb[g] = maskf;
}

// ============================================================
// weight convert: f32 [D][K][N] -> bf16 tiled [D][k/32][n(Np)][k%32]
// ============================================================
__global__ void k_wcvt(const float* __restrict__ src, unsigned short* __restrict__ dst,
                       int K, int N, int Kp, int Np, int D)
{
  long long per = (long long)Kp * Np;
  long long total = per * D;
  long long i = (long long)blockIdx.x * blockDim.x + threadIdx.x;
  long long st = (long long)gridDim.x * blockDim.x;
  for (; i < total; i += st) {
    int d = (int)(i / per);
    int rem = (int)(i - (long long)d * per);
    int k = rem / Np, n = rem - (rem / Np) * Np;
    float v = 0.f;
    if (k < K && n < N) v = src[((long long)d * K + k) * N + n];
    long long o = (long long)d * per + ((long long)(k >> 5) * Np + n) * 32 + (k & 31);
    dst[o] = f2bf(v);
  }
}

// ============================================================
// bf16 WMMA GEMM: out = act(alpha*(A@B) + bias) + res
// A: M x Kp bf16 row-major (pad zero-filled); B: tiled weights
// 64x64 tile / block, 128 threads = 4 waves; guard-free staging
// ============================================================
__global__ __launch_bounds__(128)
void k_gemm(const unsigned short* __restrict__ A,
            const unsigned short* __restrict__ Bw,
            const float* __restrict__ bias,
            const float* __restrict__ res,
            const float* __restrict__ alphaP, int alphaIdx,
            float* __restrict__ outF, unsigned short* __restrict__ outH,
            int M, int N, int Kp, int Np, int act)
{
  __shared__ __align__(16) unsigned short As[64][32];
  __shared__ __align__(16) unsigned short Bs[64][32];   // [n][k] within tile
  int tid = threadIdx.x, lane = tid & 31, wave = tid >> 5;
  int half = lane >> 4, l16 = lane & 15;
  int m0 = blockIdx.y * 64, n0 = blockIdx.x * 64;
  const unsigned short* Abase = A + (size_t)m0 * Kp;
  v8f acc[4] = {};
  for (int kb = 0; kb < Kp; kb += 32) {
    const unsigned short* Bsrc = Bw + ((size_t)(kb >> 5) * Np + n0) * 32;
#if HAVE_ASYNC_LDS
    // CDNA5 async global->LDS copy (ASYNCcnt), no VGPR round-trip
    #pragma unroll
    for (int j = 0; j < 2; ++j) {
      int e = tid + j * 128;
      int row = e >> 2, ch = (e & 3) * 8;
      __builtin_amdgcn_global_load_async_to_lds_b128(
          (gptr_v4i)(Abase + (size_t)row * Kp + kb + ch),
          (lptr_v4i)(&As[row][ch]), 0, 0);
      __builtin_amdgcn_global_load_async_to_lds_b128(
          (gptr_v4i)(Bsrc + e * 8),
          (lptr_v4i)(&Bs[0][0] + e * 8), 0, 0);
    }
#else
    #pragma unroll
    for (int j = 0; j < 2; ++j) {
      int e = tid + j * 128;
      int row = e >> 2, ch = (e & 3) * 8;
      *(uint4*)&As[row][ch] = *(const uint4*)(Abase + (size_t)row * Kp + kb + ch);
    }
    #pragma unroll
    for (int j = 0; j < 2; ++j) {
      int e = tid + j * 128;
      *(uint4*)(&Bs[0][0] + e * 8) = *(const uint4*)(Bsrc + e * 8);
    }
#endif
    // speculative prefetch of next k-step (dropped silently if OOB)
    __builtin_prefetch(Abase + (size_t)(tid >> 1) * Kp + kb + 32, 0, 1);
    __builtin_prefetch(Bsrc + (size_t)Np * 32 + tid * 16, 0, 1);
#if HAVE_ASYNC_LDS
    __builtin_amdgcn_s_wait_asynccnt(0);
#endif
    __syncthreads();
    int arow = wave * 16 + l16;
    int ks = half * 8;
    v16u au;
    #pragma unroll
    for (int i = 0; i < 8; ++i) {
      au[i]     = As[arow][ks + i];
      au[8 + i] = As[arow][16 + ks + i];
    }
    v16bf a = __builtin_bit_cast(v16bf, au);
    #pragma unroll
    for (int nt = 0; nt < 4; ++nt) {
      int nn = nt * 16 + l16;
      v16u bu;
      #pragma unroll
      for (int i = 0; i < 16; ++i) bu[i] = Bs[nn][half * 8 + (i & 7) + ((i >> 3) << 4)];
      v16bf bf_ = __builtin_bit_cast(v16bf, bu);
      acc[nt] = __builtin_amdgcn_wmma_f32_16x16x32_bf16(
          false, a, false, bf_, (short)0, acc[nt], false, false);
    }
    __syncthreads();
  }
  float alpha = alphaP ? alphaP[alphaIdx] : 1.0f;
  #pragma unroll
  for (int nt = 0; nt < 4; ++nt) {
    #pragma unroll
    for (int r = 0; r < 8; ++r) {
      int gm = m0 + wave * 16 + half * 8 + r;
      int gn = n0 + nt * 16 + l16;
      if (gm < M && gn < N) {
        float v = alpha * acc[nt][r];
        if (bias) v += bias[gn];
        if (act)  v = 0.5f * v * (1.0f + erff(v * 0.70710678118654752f));
        if (res)  v += res[(size_t)gm * N + gn];
        size_t oi = (size_t)gm * N + gn;
        if (outF) outF[oi] = v;
        if (outH) outH[oi] = f2bf(v);
      }
    }
  }
}

// ============================================================
// flash attention: block = (b, head); 256 thr = 8 waves x 16 q-rows
// Q: [B][128][600] bf16 ; KV: [B][nk][1200] bf16 ; Out stride 608
// ============================================================
__global__ __launch_bounds__(256)
void k_attn(const unsigned short* __restrict__ Q,
            const unsigned short* __restrict__ KV,
            const float* __restrict__ maskb,
            unsigned short* __restrict__ Out,
            int nk, float scale)
{
  __shared__ __align__(16) unsigned short Qs[NL][DPP];
  __shared__ __align__(16) unsigned short Ks[KT][DPP];
  __shared__ __align__(16) unsigned short Vt[DP][KT];
  __shared__ float mk[KT];
  __shared__ __align__(16) unsigned short Ps[8][16][KT];
  int b  = blockIdx.x >> 3;
  int hd = blockIdx.x & 7;
  int tid = threadIdx.x, lane = tid & 31, wave = tid >> 5;
  int half = lane >> 4, l16 = lane & 15;

  // stage Q (dense, then zero-pad cols 75..95)
  for (int i = tid; i < NL * DH; i += 256) {
    int r = i / DH, d = i - r * DH;
    Qs[r][d] = f2bf(bf2f(Q[(size_t)(b * NL + r) * INW + hd * DH + d]) * scale);
  }
  for (int i = tid; i < NL * (DPP - DH); i += 256) {
    int r = i / (DPP - DH), d = DH + (i - r * (DPP - DH));
    Qs[r][d] = 0;
  }
  __syncthreads();

  float mrow[8], lrow[8];
  v8f Oacc[5] = {};
  #pragma unroll
  for (int r = 0; r < 8; ++r) { mrow[r] = -3.0e38f; lrow[r] = 0.f; }

  for (int k0 = 0; k0 < nk; k0 += KT) {
    for (int i = tid; i < KT * DH; i += 256) {
      int kk = i / DH, d = i - kk * DH;
      size_t base = (size_t)(b * nk + k0 + kk) * KVW + hd * DH;
      Ks[kk][d] = KV[base + d];
      Vt[d][kk] = KV[base + INW + d];
    }
    for (int i = tid; i < KT * (DPP - DH); i += 256) {
      int kk = i / (DPP - DH), d = DH + (i - kk * (DPP - DH));
      Ks[kk][d] = 0;
    }
    for (int i = tid; i < KT * (DP - DH); i += 256) {
      int kk = i / (DP - DH), d = DH + (i - kk * (DP - DH));
      Vt[d][kk] = 0;
    }
    if (tid < KT) mk[tid] = maskb ? maskb[(size_t)b * nk + k0 + tid] : 1.0f;
    __syncthreads();

    // S = Q K^T : Kdim = 96, guard-free
    v8f S[4] = {};
    #pragma unroll
    for (int kb = 0; kb < DPP; kb += 32) {
      int arow = wave * 16 + l16;
      int ks = half * 8;
      v16u au;
      #pragma unroll
      for (int i = 0; i < 8; ++i) {
        au[i]     = Qs[arow][kb + ks + i];
        au[8 + i] = Qs[arow][kb + 16 + ks + i];
      }
      v16bf a = __builtin_bit_cast(v16bf, au);
      #pragma unroll
      for (int nt = 0; nt < 4; ++nt) {
        int nn = nt * 16 + l16;
        v16u bu;
        #pragma unroll
        for (int i = 0; i < 16; ++i) bu[i] = Ks[nn][kb + half * 8 + (i & 7) + ((i >> 3) << 4)];
        v16bf bf_ = __builtin_bit_cast(v16bf, bu);
        S[nt] = __builtin_amdgcn_wmma_f32_16x16x32_bf16(
            false, a, false, bf_, (short)0, S[nt], false, false);
      }
    }

    // online softmax; lane owns rows half*8+r, col l16 of each n-tile
    #pragma unroll
    for (int r = 0; r < 8; ++r) {
      float sv[4];
      float tm = -3.0e38f;
      #pragma unroll
      for (int nt = 0; nt < 4; ++nt) {
        float s = S[nt][r];
        s = (mk[nt * 16 + l16] < 0.5f) ? -3.0e38f : s;
        sv[nt] = s;
        tm = fmaxf(tm, s);
      }
      #pragma unroll
      for (int m = 8; m >= 1; m >>= 1) tm = fmaxf(tm, __shfl_xor(tm, m, 32));
      float nm = fmaxf(mrow[r], tm);
      float corr = __expf(mrow[r] - nm);
      mrow[r] = nm;
      float ts = 0.f;
      #pragma unroll
      for (int nt = 0; nt < 4; ++nt) {
        float p = __expf(sv[nt] - nm);
        ts += p;
        Ps[wave][half * 8 + r][nt * 16 + l16] = f2bf(p);
      }
      #pragma unroll
      for (int m = 8; m >= 1; m >>= 1) ts += __shfl_xor(ts, m, 32);
      lrow[r] = lrow[r] * corr + ts;
      #pragma unroll
      for (int dt = 0; dt < 5; ++dt) Oacc[dt][r] *= corr;
    }

    // O += P @ V : Kdim = 64
    #pragma unroll
    for (int kb = 0; kb < KT; kb += 32) {
      v16u au;
      #pragma unroll
      for (int i = 0; i < 8; ++i) {
        au[i]     = Ps[wave][l16][kb + half * 8 + i];
        au[8 + i] = Ps[wave][l16][kb + 16 + half * 8 + i];
      }
      v16bf a = __builtin_bit_cast(v16bf, au);
      #pragma unroll
      for (int dt = 0; dt < 5; ++dt) {
        int nn = dt * 16 + l16;
        v16u bu;
        #pragma unroll
        for (int i = 0; i < 16; ++i) bu[i] = Vt[nn][kb + half * 8 + (i & 7) + ((i >> 3) << 4)];
        v16bf bf_ = __builtin_bit_cast(v16bf, bu);
        Oacc[dt] = __builtin_amdgcn_wmma_f32_16x16x32_bf16(
            false, a, false, bf_, (short)0, Oacc[dt], false, false);
      }
    }
    __syncthreads();
  }

  #pragma unroll
  for (int dt = 0; dt < 5; ++dt) {
    #pragma unroll
    for (int r = 0; r < 8; ++r) {
      int d = dt * 16 + l16;
      int row = wave * 16 + half * 8 + r;
      if (d < DH) {
        float denom = lrow[r];
        denom = (denom > 0.f) ? denom : 1.f;
        Out[(size_t)(b * NL + row) * INWP + hd * DH + d] = f2bf(Oacc[dt][r] / denom);
      }
    }
  }
}

// ============================================================
// small utility kernels
// ============================================================
__global__ __launch_bounds__(256)
void k_rownorm(const float* __restrict__ X, unsigned short* __restrict__ Y,
               int rows, int cols, int do_norm)
{
  int lane = threadIdx.x & 31, wave = threadIdx.x >> 5;
  int row = blockIdx.x * 8 + wave;
  if (row >= rows) return;
  const float* xr = X + (size_t)row * cols;
  float ss = 0.f;
  for (int c = lane; c < cols; c += 32) { float v = xr[c]; ss += v * v; }
  #pragma unroll
  for (int m = 16; m >= 1; m >>= 1) ss += __shfl_xor(ss, m, 32);
  float inv = do_norm ? (1.f / fmaxf(sqrtf(ss), 1e-5f)) : 1.f;
  unsigned short* yr = Y + (size_t)row * cols;
  for (int c = lane; c < cols; c += 32) yr[c] = f2bf(xr[c] * inv);
}

__global__ void k_latinit(const float* __restrict__ lat0, const float* __restrict__ pe,
                          float* __restrict__ lat, int per_b, int total)
{
  int i = blockIdx.x * blockDim.x + threadIdx.x;
  if (i < total) { int j = i % per_b; lat[i] = lat0[j] + pe[j]; }
}

__global__ void k_reorder(const float* __restrict__ lg, float* __restrict__ out)
{
  int i = blockIdx.x * 256 + threadIdx.x;       // NB*NL*OCW = 32768
  if (i < NB * NL * OCW) {
    int oc = i % OCW;
    int l = (i / OCW) % NL;
    int b = i / (OCW * NL);
    int t = l % NTB, nn = l / NTB;
    out[(((size_t)t * NB + b) * (NL / NTB) + nn) * OCW + oc] = lg[i];
  }
}

// ============================================================
// host orchestration
// ============================================================
extern "C" void kernel_launch(void* const* d_in, const int* in_sizes, int n_in,
                              void* d_out, int out_size, void* d_ws, size_t ws_size,
                              hipStream_t stream)
{
  const float* g_input   = (const float*)d_in[0];
  const float* g_latents = (const float*)d_in[1];
  const float* g_posemb  = (const float*)d_in[2];
  const float* g_ca_g    = (const float*)d_in[3];
  const float* g_ca_ctxg = (const float*)d_in[4];
  const float* g_ca_q    = (const float*)d_in[5];
  const float* g_ca_kv   = (const float*)d_in[6];
  const float* g_ca_ow   = (const float*)d_in[7];
  const float* g_ca_ob   = (const float*)d_in[8];
  const float* g_cf_g    = (const float*)d_in[9];
  const float* g_cf_w1   = (const float*)d_in[10];
  const float* g_cf_b1   = (const float*)d_in[11];
  const float* g_cf_w2   = (const float*)d_in[12];
  const float* g_cf_b2   = (const float*)d_in[13];
  const float* g_la_g    = (const float*)d_in[14];
  const float* g_la_q    = (const float*)d_in[15];
  const float* g_la_kv   = (const float*)d_in[16];
  const float* g_la_ow   = (const float*)d_in[17];
  const float* g_la_ob   = (const float*)d_in[18];
  const float* g_lf_g    = (const float*)d_in[19];
  const float* g_lf_w1   = (const float*)d_in[20];
  const float* g_lf_b1   = (const float*)d_in[21];
  const float* g_lf_w2   = (const float*)d_in[22];
  const float* g_lf_b2   = (const float*)d_in[23];
  const float* g_logw    = (const float*)d_in[24];
  const float* g_logb    = (const float*)d_in[25];

  char* ws = (char*)d_ws;
  size_t off = 0;
  auto take = [&](size_t bytes) -> void* {
    void* p = ws + off;
    off = (off + bytes + 255) & ~(size_t)255;
    return p;
  };

  const size_t n_caq  = (size_t)DEPTH * LD   * INWN;
  const size_t n_cakv = (size_t)DEPTH * IDWP * KVWN;
  const size_t n_caow = (size_t)DEPTH * INWP * LD;
  const size_t n_w1   = (size_t)DEPTH * LD   * FFW;
  const size_t n_w2   = (size_t)DEPTH * FFW  * LD;
  const size_t n_lakv = (size_t)DEPTH * LD   * KVWN;
  const size_t n_logw = (size_t)LD * OCW;

  unsigned short* b_caq  = (unsigned short*)take(n_caq * 2);
  unsigned short* b_cakv = (unsigned short*)take(n_cakv * 2);
  unsigned short* b_caow = (unsigned short*)take(n_caow * 2);
  unsigned short* b_cfw1 = (unsigned short*)take(n_w1 * 2);
  unsigned short* b_cfw2 = (unsigned short*)take(n_w2 * 2);
  unsigned short* b_laq  = (unsigned short*)take(n_caq * 2);
  unsigned short* b_lakv = (unsigned short*)take(n_lakv * 2);
  unsigned short* b_laow = (unsigned short*)take(n_caow * 2);
  unsigned short* b_lfw1 = (unsigned short*)take(n_w1 * 2);
  unsigned short* b_lfw2 = (unsigned short*)take(n_w2 * 2);
  unsigned short* b_logw = (unsigned short*)take(n_logw * 2);

  unsigned short* dataN  = (unsigned short*)take((size_t)NB * NVOX * IDWP * 2);
  float*          maskb  = (float*)take((size_t)NB * NVOX * 4);
  float*          lat    = (float*)take((size_t)NB * NL * LD * 4);
  unsigned short* latn   = (unsigned short*)take((size_t)NB * NL * LD * 2);
  unsigned short* qbuf   = (unsigned short*)take((size_t)NB * NL * INW * 2);
  unsigned short* kvbuf  = (unsigned short*)take((size_t)NB * NVOX * KVW * 2);
  unsigned short* aout   = (unsigned short*)take((size_t)NB * NL * INWP * 2);
  unsigned short* ff1    = (unsigned short*)take((size_t)NB * NL * FFW * 2);
  float*          logits = (float*)take((size_t)NB * NL * OCW * 4);

  auto wcvt = [&](const float* s, unsigned short* d, int K, int N, int Kp, int Np, int D) {
    k_wcvt<<<4096, 256, 0, stream>>>(s, d, K, N, Kp, Np, D);
  };
  wcvt(g_ca_q,  b_caq,  LD,   INW, LD,   INWN, DEPTH);
  wcvt(g_ca_kv, b_cakv, IDW,  KVW, IDWP, KVWN, DEPTH);
  wcvt(g_ca_ow, b_caow, INW,  LD,  INWP, LD,   DEPTH);
  wcvt(g_cf_w1, b_cfw1, LD,   FFW, LD,   FFW,  DEPTH);
  wcvt(g_cf_w2, b_cfw2, FFW,  LD,  FFW,  LD,   DEPTH);
  wcvt(g_la_q,  b_laq,  LD,   INW, LD,   INWN, DEPTH);
  wcvt(g_la_kv, b_lakv, LD,   KVW, LD,   KVWN, DEPTH);
  wcvt(g_la_ow, b_laow, INW,  LD,  INWP, LD,   DEPTH);
  wcvt(g_lf_w1, b_lfw1, LD,   FFW, LD,   FFW,  DEPTH);
  wcvt(g_lf_w2, b_lfw2, FFW,  LD,  FFW,  LD,   DEPTH);
  wcvt(g_logw,  b_logw, LD,   OCW, LD,   OCW,  1);

  (void)hipMemsetAsync(dataN, 0, (size_t)NB * NVOX * IDWP * 2, stream);
  (void)hipMemsetAsync(aout,  0, (size_t)NB * NL * INWP * 2, stream);

  k_build_data<<<NB * NVOX / 4, 128, 0, stream>>>(g_input, dataN, maskb);
  k_latinit<<<(NB * NL * LD + 255) / 256, 256, 0, stream>>>(
      g_latents, g_posemb, lat, NL * LD, NB * NL * LD);

  auto gemm = [&](const unsigned short* A, const unsigned short* B,
                  const float* bias, const float* res,
                  const float* alphaP, int aIdx,
                  float* oF, unsigned short* oH,
                  int M, int N, int Kp, int Np, int act) {
    dim3 grid(Np / 64, M / 64);
    k_gemm<<<grid, 128, 0, stream>>>(A, B, bias, res, alphaP, aIdx, oF, oH,
                                     M, N, Kp, Np, act);
  };

  const int MR = NB * NL;   // 512
  const float SCALE = 0.11547005383792516f;  // 75^-0.5

  for (int d = 0; d < DEPTH; ++d) {
    // ---- cross attention ----
    k_rownorm<<<MR / 8, 256, 0, stream>>>(lat, latn, MR, LD, 1);
    gemm(latn, b_caq + (size_t)d * LD * INWN, nullptr, nullptr, g_ca_g, d,
         nullptr, qbuf, MR, INW, LD, INWN, 0);
    gemm(dataN, b_cakv + (size_t)d * IDWP * KVWN, nullptr, nullptr, g_ca_ctxg, d,
         nullptr, kvbuf, NB * NVOX, KVW, IDWP, KVWN, 0);
    k_attn<<<NB * NHEAD, 256, 0, stream>>>(qbuf, kvbuf, maskb, aout, NVOX, SCALE);
    gemm(aout, b_caow + (size_t)d * INWP * LD, g_ca_ob + (size_t)d * LD, lat,
         nullptr, 0, lat, nullptr, MR, LD, INWP, LD, 0);
    // ---- FF ----
    k_rownorm<<<MR / 8, 256, 0, stream>>>(lat, latn, MR, LD, 1);
    gemm(latn, b_cfw1 + (size_t)d * LD * FFW, g_cf_b1 + (size_t)d * FFW, nullptr,
         g_cf_g, d, nullptr, ff1, MR, FFW, LD, FFW, 1);
    gemm(ff1, b_cfw2 + (size_t)d * FFW * LD, g_cf_b2 + (size_t)d * LD, lat,
         nullptr, 0, lat, nullptr, MR, LD, FFW, LD, 0);
    // ---- latent self attention ----
    k_rownorm<<<MR / 8, 256, 0, stream>>>(lat, latn, MR, LD, 1);
    gemm(latn, b_laq + (size_t)d * LD * INWN, nullptr, nullptr, g_la_g, d,
         nullptr, qbuf, MR, INW, LD, INWN, 0);
    gemm(latn, b_lakv + (size_t)d * LD * KVWN, nullptr, nullptr, g_la_g, d,
         nullptr, kvbuf, MR, KVW, LD, KVWN, 0);
    k_attn<<<NB * NHEAD, 256, 0, stream>>>(qbuf, kvbuf, nullptr, aout, NL, SCALE);
    gemm(aout, b_laow + (size_t)d * INWP * LD, g_la_ob + (size_t)d * LD, lat,
         nullptr, 0, lat, nullptr, MR, LD, INWP, LD, 0);
    // ---- FF ----
    k_rownorm<<<MR / 8, 256, 0, stream>>>(lat, latn, MR, LD, 1);
    gemm(latn, b_lfw1 + (size_t)d * LD * FFW, g_lf_b1 + (size_t)d * FFW, nullptr,
         g_lf_g, d, nullptr, ff1, MR, FFW, LD, FFW, 1);
    gemm(ff1, b_lfw2 + (size_t)d * FFW * LD, g_lf_b2 + (size_t)d * LD, lat,
         nullptr, 0, lat, nullptr, MR, LD, FFW, LD, 0);
  }

  // logits head + output reorder (b,128,64) -> (t,b,16,64)
  k_rownorm<<<MR / 8, 256, 0, stream>>>(lat, latn, MR, LD, 0);
  gemm(latn, b_logw, g_logb, nullptr, nullptr, 0, logits, nullptr,
       MR, OCW, LD, OCW, 0);
  k_reorder<<<(NB * NL * OCW + 255) / 256, 256, 0, stream>>>(logits, (float*)d_out);

  (void)in_sizes; (void)n_in; (void)out_size; (void)ws_size;
}
